// T5BlockPipe_56521769615560
// MI455X (gfx1250) — compile-verified, matrix operand order checked by reading, not run on hardware
//
#include <hip/hip_runtime.h>

// ---------------------------------------------------------------------------
// T5 encoder block for MI455X (gfx1250, wave32, WMMA).
// bf16 WMMA (f32 accum) for all GEMMs; flash-style attention; async global->LDS
// staging (guarded) for the GEMM pipeline; pos_bias generated from a delta table.
// ---------------------------------------------------------------------------

typedef __attribute__((ext_vector_type(16))) __bf16 bf16x16;
typedef __attribute__((ext_vector_type(8)))  __bf16 bf16x8;
typedef __attribute__((ext_vector_type(4)))  __bf16 bf16x4;
typedef __attribute__((ext_vector_type(8)))  float  fx8;
typedef __attribute__((ext_vector_type(4)))  float  fx4;
typedef int v4i __attribute__((vector_size(16)));

#define DEV __device__ __forceinline__

// ---- async global->LDS (gfx1250) with compile-safe fallback ----------------
#if defined(__HIP_DEVICE_COMPILE__) && __has_builtin(__builtin_amdgcn_global_load_async_to_lds_b128)
#define HAVE_ASYNC_LDS 1
#endif

#if defined(__HIP_DEVICE_COMPILE__) && __has_builtin(__builtin_amdgcn_s_wait_asynccnt)
#define WAIT_ASYNC() __builtin_amdgcn_s_wait_asynccnt(0)
#elif defined(HAVE_ASYNC_LDS)
#define WAIT_ASYNC() asm volatile("s_wait_asynccnt 0" ::: "memory")
#else
#define WAIT_ASYNC() ((void)0)
#endif

DEV void copy16_g2l(__bf16* lds, const __bf16* g) {
#ifdef HAVE_ASYNC_LDS
  typedef __attribute__((address_space(1))) v4i GV4;  // global int4
  typedef __attribute__((address_space(3))) v4i LV4;  // LDS int4
  __builtin_amdgcn_global_load_async_to_lds_b128((GV4*)g, (LV4*)lds, 0, 0);
#else
  *(fx4*)lds = *(const fx4*)g;
#endif
}

// Load a 16x32 bf16 A/B fragment row for one lane. `row` points at (m, k=0);
// lane's 16 values are cols [hf*8, hf*8+7] and [16+hf*8, 16+hf*8+7]  (ISA 7.12.2).
DEV bf16x16 load_frag(const __bf16* row, int hf) {
  bf16x8 lo = *(const bf16x8*)(row + hf * 8);
  bf16x8 hi = *(const bf16x8*)(row + 16 + hf * 8);
  return __builtin_shufflevector(lo, hi, 0,1,2,3,4,5,6,7,8,9,10,11,12,13,14,15);
}

DEV fx8 wmma_bf16(bf16x16 a, bf16x16 b, fx8 c) {
  return __builtin_amdgcn_wmma_f32_16x16x32_bf16(false, a, false, b, (short)0, c,
                                                 false, false);
}

// ---------------------------------------------------------------------------
// Weight cast + transpose: src f32 [K][N] -> dst bf16 [N][K]
// ---------------------------------------------------------------------------
__launch_bounds__(256)
__global__ void cast_transpose_k(const float* __restrict__ src,
                                 __bf16* __restrict__ dst, int K, int N) {
  __shared__ float tile[32][33];
  const int k0 = blockIdx.y * 32, n0 = blockIdx.x * 32;
  const int tx = threadIdx.x & 31, ty = threadIdx.x >> 5;  // 8 rows / pass
#pragma unroll
  for (int p = 0; p < 4; ++p)
    tile[ty + p * 8][tx] = src[(size_t)(k0 + ty + p * 8) * N + n0 + tx];
  __syncthreads();
#pragma unroll
  for (int p = 0; p < 4; ++p)
    dst[(size_t)(n0 + ty + p * 8) * K + k0 + tx] = (__bf16)tile[tx][ty + p * 8];
}

// ---------------------------------------------------------------------------
// RMSNorm: f32 [rows][1024] * w -> bf16, one row per 256-thread block
// ---------------------------------------------------------------------------
__launch_bounds__(256)
__global__ void rmsnorm_bf16_k(const float* __restrict__ x,
                               const float* __restrict__ w,
                               __bf16* __restrict__ out) {
  const int row = blockIdx.x, t = threadIdx.x;
  const int lane = t & 31, wid = t >> 5;
  const float* xr = x + (size_t)row * 1024;
  fx4 xv = *(const fx4*)(xr + t * 4);
  float ss = xv[0]*xv[0] + xv[1]*xv[1] + xv[2]*xv[2] + xv[3]*xv[3];
#pragma unroll
  for (int off = 16; off > 0; off >>= 1) ss += __shfl_xor(ss, off, 32);
  __shared__ float red[8];
  if (lane == 0) red[wid] = ss;
  __syncthreads();
  float tot = 0.f;
#pragma unroll
  for (int i = 0; i < 8; ++i) tot += red[i];
  const float rs = rsqrtf(tot * (1.0f / 1024.0f) + 1e-6f);
  fx4 wv = *(const fx4*)(w + t * 4);
  bf16x4 o;
#pragma unroll
  for (int i = 0; i < 4; ++i) o[i] = (__bf16)(xv[i] * rs * wv[i]);
  *(bf16x4*)(out + (size_t)row * 1024 + t * 4) = o;
}

// ---------------------------------------------------------------------------
// bf16 WMMA GEMM: C[M,N] = A[M,K] @ Bt[N,K]^T
// MODE 0: bf16 C; MODE 1: f32 C = resid + acc; MODE 2: bf16 C = relu(acc)
// Block tile 128x128 (8 waves, 64x32 each), K-step 32, double-buffered LDS.
// ---------------------------------------------------------------------------
template <int MODE>
__launch_bounds__(256, 2)
__global__ void gemm_bf16_wmma(const __bf16* __restrict__ A,
                               const __bf16* __restrict__ Bt,
                               void* __restrict__ Cout,
                               const float* __restrict__ resid,
                               int M, int N, int K) {
  constexpr int TM = 128, TN = 128, TK = 32, LDT = 40;  // 16B row pad
  __shared__ __align__(16) __bf16 sA[2][TM * LDT];
  __shared__ __align__(16) __bf16 sB[2][TN * LDT];

  const int t = threadIdx.x;
  const int lane = t & 31, wid = t >> 5;
  const int l16 = lane & 15, hf = lane >> 4;
  const int wm = wid & 1, wn = wid >> 1;
  const long m0 = (long)blockIdx.y * TM;
  const long n0 = (long)blockIdx.x * TN;

  fx8 zero = {};
  fx8 acc[4][2];
#pragma unroll
  for (int i = 0; i < 4; ++i)
#pragma unroll
    for (int j = 0; j < 2; ++j) acc[i][j] = zero;

  auto stage_tile = [&](int buf, int kt) {
    const long k0 = (long)kt * TK;
#pragma unroll
    for (int p = 0; p < 2; ++p) {
      const int c = t + p * 256;
      const int row = c >> 2, cj = c & 3;
      copy16_g2l(&sA[buf][row * LDT + cj * 8],
                 A + (m0 + row) * K + k0 + cj * 8);
    }
#pragma unroll
    for (int p = 0; p < 2; ++p) {
      const int c = t + p * 256;
      const int row = c >> 2, cj = c & 3;
      copy16_g2l(&sB[buf][row * LDT + cj * 8],
                 Bt + (n0 + row) * K + k0 + cj * 8);
    }
  };

  const int nk = K / TK;
  stage_tile(0, 0);
  WAIT_ASYNC();
  __syncthreads();

  for (int kt = 0; kt < nk; ++kt) {
    const int buf = kt & 1;
    if (kt + 1 < nk) stage_tile(buf ^ 1, kt + 1);

    bf16x16 af[4], bfr[2];
#pragma unroll
    for (int mi = 0; mi < 4; ++mi)
      af[mi] = load_frag(&sA[buf][(wm * 64 + mi * 16 + l16) * LDT], hf);
#pragma unroll
    for (int ni = 0; ni < 2; ++ni)
      bfr[ni] = load_frag(&sB[buf][(wn * 32 + ni * 16 + l16) * LDT], hf);

#pragma unroll
    for (int mi = 0; mi < 4; ++mi)
#pragma unroll
      for (int ni = 0; ni < 2; ++ni)
        acc[mi][ni] = wmma_bf16(af[mi], bfr[ni], acc[mi][ni]);

    WAIT_ASYNC();
    __syncthreads();
  }

#pragma unroll
  for (int mi = 0; mi < 4; ++mi) {
#pragma unroll
    for (int ni = 0; ni < 2; ++ni) {
      const long n = n0 + wn * 32 + ni * 16 + l16;
#pragma unroll
      for (int r = 0; r < 8; ++r) {
        const long m = m0 + wm * 64 + mi * 16 + hf * 8 + r;
        const size_t idx = (size_t)m * N + n;
        const float v = acc[mi][ni][r];
        if (MODE == 0)      ((__bf16*)Cout)[idx] = (__bf16)v;
        else if (MODE == 1) ((float*)Cout)[idx] = resid[idx] + v;
        else                ((__bf16*)Cout)[idx] = (__bf16)fmaxf(v, 0.0f);
      }
    }
  }
}

// ---------------------------------------------------------------------------
// V transpose: v[b,s,h,d] bf16 -> vt[b,h,d,s] bf16
// ---------------------------------------------------------------------------
__launch_bounds__(256)
__global__ void transpose_v_k(const __bf16* __restrict__ v,
                              __bf16* __restrict__ vt) {
  __shared__ __bf16 tile[64][72];
  const int s0 = blockIdx.x * 64, h = blockIdx.y, b = blockIdx.z;
  const int tx = threadIdx.x & 63, ty = threadIdx.x >> 6;  // 4 rows / pass
#pragma unroll
  for (int p = 0; p < 16; ++p) {
    const int sl = ty + p * 4;
    tile[sl][tx] = v[((size_t)(b * 1024 + s0 + sl) * 1024) + h * 64 + tx];
  }
  __syncthreads();
#pragma unroll
  for (int p = 0; p < 16; ++p) {
    const int d = ty + p * 4;
    vt[((size_t)((b * 16 + h) * 64 + d) * 1024) + s0 + tx] = tile[tx][d];
  }
}

// ---------------------------------------------------------------------------
// Relative-position bias tables.
// biastab[h][delta] = rel_emb[bucket(delta-1023)][h], delta in [0,2046]
// maskneg[b][k] = (1 - mask[b][k]) * -1e9
// ---------------------------------------------------------------------------
DEV int rel_bucket(int rel) {
  const int nb = 16, max_exact = 8;
  const int rb = (rel > 0) ? nb : 0;
  int rp = rel < 0 ? -rel : rel;
  if (rp < max_exact) return rb + rp;
  const float rp_f = (float)(rp < 1 ? 1 : rp);
  int large = max_exact +
              (int)(logf(rp_f / 8.0f) / logf(128.0f / 8.0f) * 8.0f);
  large = large < nb - 1 ? large : nb - 1;
  return rb + large;
}

__launch_bounds__(256)
__global__ void build_tables_k(const float* __restrict__ rel_emb,
                               const float* __restrict__ mask,
                               float* __restrict__ biastab,
                               float* __restrict__ maskneg) {
  const int t = blockIdx.x * blockDim.x + threadIdx.x;
  const int stride = gridDim.x * blockDim.x;
  for (int i = t; i < 16 * 2047; i += stride) {
    const int h = i / 2047, d = i % 2047;
    const int bucket = rel_bucket(d - 1023);
    biastab[i] = rel_emb[bucket * 16 + h];
  }
  for (int i = t; i < 8 * 1024; i += stride)
    maskneg[i] = (1.0f - mask[i]) * -1e9f;
}

// pos_bias[b,h,q,k] = biastab[h][k-q+1023] + maskneg[b][k]
__launch_bounds__(256)
__global__ void posbias_out_k(const float* __restrict__ biastab,
                              const float* __restrict__ maskneg,
                              float* __restrict__ out) {
  const int qq = blockIdx.x, h = blockIdx.y, b = blockIdx.z;
  const int t = threadIdx.x;
  const float* bt = biastab + h * 2047 + (1023 - qq);
  const float* mk = maskneg + b * 1024;
  float* o = out + (((size_t)(b * 16 + h) * 1024 + qq) * 1024);
  const int k = t * 4;
  fx4 v;
#pragma unroll
  for (int j = 0; j < 4; ++j) v[j] = bt[k + j] + mk[k + j];
  *(fx4*)(o + k) = v;
}

// ---------------------------------------------------------------------------
// Flash attention. Block = (b, h, 128 q-rows); 8 waves, 16 q-rows each.
// scores = Q@K^T + bias, streamed softmax, ctx = P@V (Vt pre-transposed).
// ---------------------------------------------------------------------------
__launch_bounds__(256, 2)
__global__ void flash_attn_k(const __bf16* __restrict__ q,
                             const __bf16* __restrict__ kk,
                             const __bf16* __restrict__ vt,
                             const float* __restrict__ biastab,
                             const float* __restrict__ maskneg,
                             __bf16* __restrict__ ctx) {
  constexpr int S = 1024, H = 16, DK = 64, LDP = 40;
  __shared__ float sBias[2047];
  __shared__ float sMask[1024];
  __shared__ __align__(16) __bf16 sP[8][16 * LDP];

  const int t = threadIdx.x, lane = t & 31, wid = t >> 5;
  const int l16 = lane & 15, hf = lane >> 4;
  const int b = blockIdx.z, h = blockIdx.y;
  const int q0 = blockIdx.x * 128 + wid * 16;

  for (int i = t; i < 2047; i += 256) sBias[i] = biastab[h * 2047 + i];
  for (int i = t; i < 1024; i += 256) sMask[i] = maskneg[b * 1024 + i];
  __syncthreads();

  const __bf16* qrow = q + ((size_t)(b * S + q0 + l16) * (H * DK)) + h * DK;
  const bf16x16 aq0 = load_frag(qrow, hf);       // d 0..31
  const bf16x16 aq1 = load_frag(qrow + 32, hf);  // d 32..63

  float mrun[8], lrun[8];
#pragma unroll
  for (int r = 0; r < 8; ++r) { mrun[r] = -1e30f; lrun[r] = 0.0f; }
  fx8 zero = {};
  fx8 accv[4];
#pragma unroll
  for (int ns = 0; ns < 4; ++ns) accv[ns] = zero;

  const __bf16* kbase = kk + (size_t)b * S * (H * DK) + h * DK;
  const __bf16* vbase = vt + (size_t)((b * H + h) * DK) * S;

  for (int k0 = 0; k0 < S; k0 += 32) {
    // ---- scores: two 16x16 tiles over k columns [k0,k0+15], [k0+16,k0+31]
    fx8 s0 = zero, s1 = zero;
    {
      const __bf16* krow = kbase + (size_t)(k0 + l16) * (H * DK);
      s0 = wmma_bf16(aq0, load_frag(krow, hf), s0);
      s0 = wmma_bf16(aq1, load_frag(krow + 32, hf), s0);
      const __bf16* krow1 = kbase + (size_t)(k0 + 16 + l16) * (H * DK);
      s1 = wmma_bf16(aq0, load_frag(krow1, hf), s1);
      s1 = wmma_bf16(aq1, load_frag(krow1 + 32, hf), s1);
    }
    const int kc0 = k0 + l16, kc1 = k0 + 16 + l16;
    const float m0b = sMask[kc0], m1b = sMask[kc1];
#pragma unroll
    for (int r = 0; r < 8; ++r) {
      const int qg = q0 + hf * 8 + r;
      s0[r] += sBias[kc0 - qg + 1023] + m0b;
      s1[r] += sBias[kc1 - qg + 1023] + m1b;
    }

    // ---- online softmax (row m = hf*8 + r lives across the 16-lane half)
    float rmax[8];
#pragma unroll
    for (int r = 0; r < 8; ++r) rmax[r] = fmaxf(s0[r], s1[r]);
#pragma unroll
    for (int off = 1; off < 16; off <<= 1)
#pragma unroll
      for (int r = 0; r < 8; ++r)
        rmax[r] = fmaxf(rmax[r], __shfl_xor(rmax[r], off, 32));

    float scl[8], rsum[8];
#pragma unroll
    for (int r = 0; r < 8; ++r) {
      const float mnew = fmaxf(mrun[r], rmax[r]);
      scl[r] = __expf(mrun[r] - mnew);
      const float p0 = __expf(s0[r] - mnew);
      const float p1 = __expf(s1[r] - mnew);
      rsum[r] = p0 + p1;
      mrun[r] = mnew;
      const int m = hf * 8 + r;
      sP[wid][m * LDP + l16]      = (__bf16)p0;
      sP[wid][m * LDP + 16 + l16] = (__bf16)p1;
    }
#pragma unroll
    for (int off = 1; off < 16; off <<= 1)
#pragma unroll
      for (int r = 0; r < 8; ++r) rsum[r] += __shfl_xor(rsum[r], off, 32);
#pragma unroll
    for (int r = 0; r < 8; ++r) lrun[r] = lrun[r] * scl[r] + rsum[r];
#pragma unroll
    for (int ns = 0; ns < 4; ++ns)
#pragma unroll
      for (int r = 0; r < 8; ++r) accv[ns][r] *= scl[r];

    // ---- ctx += P(16x32) @ V(32x64)
    const bf16x16 pf = load_frag(&sP[wid][l16 * LDP], hf);
#pragma unroll
    for (int ns = 0; ns < 4; ++ns) {
      const __bf16* vrow = vbase + (size_t)(ns * 16 + l16) * S + k0;
      accv[ns] = wmma_bf16(pf, load_frag(vrow, hf), accv[ns]);
    }
  }

#pragma unroll
  for (int ns = 0; ns < 4; ++ns)
#pragma unroll
    for (int r = 0; r < 8; ++r) {
      const int m = q0 + hf * 8 + r;
      ctx[((size_t)(b * S + m) * (H * DK)) + h * DK + ns * 16 + l16] =
          (__bf16)(accv[ns][r] / lrun[r]);
    }
}

// ---------------------------------------------------------------------------
// Host launch
// ---------------------------------------------------------------------------
extern "C" void kernel_launch(void* const* d_in, const int* in_sizes, int n_in,
                              void* d_out, int out_size, void* d_ws,
                              size_t ws_size, hipStream_t stream) {
  (void)in_sizes; (void)n_in; (void)out_size; (void)ws_size;
  const float* hidden = (const float*)d_in[0];
  const float* mask   = (const float*)d_in[1];
  const float* ln1    = (const float*)d_in[3];
  const float* qw     = (const float*)d_in[4];
  const float* kw     = (const float*)d_in[5];
  const float* vw     = (const float*)d_in[6];
  const float* ow     = (const float*)d_in[7];
  const float* rel    = (const float*)d_in[8];
  const float* ln2    = (const float*)d_in[9];
  const float* wi     = (const float*)d_in[10];
  const float* wo     = (const float*)d_in[11];

  char* ws = (char*)d_ws;
  constexpr size_t SZ_W1  = 1024ull * 1024 * 2;   // 2 MiB  (1024x1024 bf16)
  constexpr size_t SZ_WI  = 4096ull * 1024 * 2;   // 8 MiB
  constexpr size_t SZ_ACT1= 8192ull * 1024 * 2;   // 16 MiB (8192x1024 bf16)
  constexpr size_t OFF_WQT  = 0;
  constexpr size_t OFF_WKT  = OFF_WQT  + SZ_W1;
  constexpr size_t OFF_WVT  = OFF_WKT  + SZ_W1;
  constexpr size_t OFF_WOT  = OFF_WVT  + SZ_W1;
  constexpr size_t OFF_WIT  = OFF_WOT  + SZ_W1;
  constexpr size_t OFF_WO2T = OFF_WIT  + SZ_WI;
  constexpr size_t OFF_XLN  = OFF_WO2T + SZ_WI;
  constexpr size_t OFF_Q    = OFF_XLN  + SZ_ACT1;
  constexpr size_t OFF_K    = OFF_Q    + SZ_ACT1;
  constexpr size_t OFF_V    = OFF_K    + SZ_ACT1;
  constexpr size_t OFF_VT   = OFF_V    + SZ_ACT1;
  constexpr size_t OFF_CTX  = OFF_VT   + SZ_ACT1;
  constexpr size_t OFF_H1   = OFF_CTX  + SZ_ACT1;           // f32, 32 MiB
  constexpr size_t OFF_ACT  = OFF_H1   + 8192ull*1024*4;    // bf16, 64 MiB
  constexpr size_t OFF_BT   = OFF_ACT  + 8192ull*4096*2;    // 16*2047 f32
  constexpr size_t OFF_MN   = OFF_BT   + 131072;            // 8*1024 f32

  __bf16* WQT  = (__bf16*)(ws + OFF_WQT);
  __bf16* WKT  = (__bf16*)(ws + OFF_WKT);
  __bf16* WVT  = (__bf16*)(ws + OFF_WVT);
  __bf16* WOT  = (__bf16*)(ws + OFF_WOT);
  __bf16* WIT  = (__bf16*)(ws + OFF_WIT);
  __bf16* WO2T = (__bf16*)(ws + OFF_WO2T);
  __bf16* XLN  = (__bf16*)(ws + OFF_XLN);
  __bf16* Q    = (__bf16*)(ws + OFF_Q);
  __bf16* K    = (__bf16*)(ws + OFF_K);
  __bf16* V    = (__bf16*)(ws + OFF_V);
  __bf16* VT   = (__bf16*)(ws + OFF_VT);
  __bf16* CTX  = (__bf16*)(ws + OFF_CTX);
  float*  H1   = (float*)(ws + OFF_H1);
  __bf16* ACT  = (__bf16*)(ws + OFF_ACT);
  float*  BT   = (float*)(ws + OFF_BT);
  float*  MN   = (float*)(ws + OFF_MN);

  float* out_h    = (float*)d_out;
  float* out_bias = out_h + 8ull * 1024 * 1024;

  const dim3 blk(256);

  // 1) weight casts/transposes (stay L2-resident afterwards)
  cast_transpose_k<<<dim3(32, 32),  blk, 0, stream>>>(qw, WQT, 1024, 1024);
  cast_transpose_k<<<dim3(32, 32),  blk, 0, stream>>>(kw, WKT, 1024, 1024);
  cast_transpose_k<<<dim3(32, 32),  blk, 0, stream>>>(vw, WVT, 1024, 1024);
  cast_transpose_k<<<dim3(32, 32),  blk, 0, stream>>>(ow, WOT, 1024, 1024);
  cast_transpose_k<<<dim3(128, 32), blk, 0, stream>>>(wi, WIT, 1024, 4096);
  cast_transpose_k<<<dim3(32, 128), blk, 0, stream>>>(wo, WO2T, 4096, 1024);

  // 2) bias tables + pos_bias output (537 MB stream store)
  build_tables_k<<<dim3(64), blk, 0, stream>>>(rel, mask, BT, MN);
  posbias_out_k<<<dim3(1024, 16, 8), blk, 0, stream>>>(BT, MN, out_bias);

  // 3) x = rmsnorm(hidden, ln1) -> bf16
  rmsnorm_bf16_k<<<dim3(8192), blk, 0, stream>>>(hidden, ln1, XLN);

  // 4) q/k/v projections
  gemm_bf16_wmma<0><<<dim3(8, 64), blk, 0, stream>>>(XLN, WQT, Q, nullptr, 8192, 1024, 1024);
  gemm_bf16_wmma<0><<<dim3(8, 64), blk, 0, stream>>>(XLN, WKT, K, nullptr, 8192, 1024, 1024);
  gemm_bf16_wmma<0><<<dim3(8, 64), blk, 0, stream>>>(XLN, WVT, V, nullptr, 8192, 1024, 1024);
  transpose_v_k<<<dim3(16, 16, 8), blk, 0, stream>>>(V, VT);

  // 5) attention
  flash_attn_k<<<dim3(8, 16, 8), blk, 0, stream>>>(Q, K, VT, BT, MN, CTX);

  // 6) h1 = hidden + ctx @ Wo
  gemm_bf16_wmma<1><<<dim3(8, 64), blk, 0, stream>>>(CTX, WOT, H1, hidden, 8192, 1024, 1024);

  // 7) FFN: y = rmsnorm(h1, ln2); act = relu(y@Wi); out = h1 + act@Wo2
  rmsnorm_bf16_k<<<dim3(8192), blk, 0, stream>>>(H1, ln2, XLN);
  gemm_bf16_wmma<2><<<dim3(32, 64), blk, 0, stream>>>(XLN, WIT, ACT, nullptr, 8192, 4096, 1024);
  gemm_bf16_wmma<1><<<dim3(8, 64), blk, 0, stream>>>(ACT, WO2T, out_h, H1, 8192, 1024, 4096);
}
